// SelfAttention_7559142441549
// MI455X (gfx1250) — compile-verified, matrix operand order checked by reading, not run on hardware
//
#include <hip/hip_runtime.h>
#include <hip/hip_bf16.h>
#include <math.h>
#include <stdint.h>

// ---------------------------------------------------------------------------
// GPT-2 self-attention block for MI455X (gfx1250), wave32 + WMMA + TDM.
//   B=2, T=4096, C=768, H=12, D=64, 3C=2304
// Pipeline:
//   1) qkv_kernel : QKV = x @ W_attn^T + b_attn  (WMMA f16->f32), writes
//                   Qh [B,H,T,D] (pre-scaled 1/sqrt(D)), Kh [B,H,T,D],
//                   Vt [B,H,D,T]  -- all f16 in workspace.
//   2) attn_kernel: flash-style softmax(Q K^T) V. One block = 8 query tiles
//                   of one (b,h); K/V tiles staged into LDS by the Tensor
//                   Data Mover (tensor_load_to_lds, TENSORcnt) and shared by
//                   all 8 waves.  Writes Y [B,T,C] f16.
//   3) proj_kernel: out = Y @ W_proj^T + b_proj (WMMA), f32 output.
// ---------------------------------------------------------------------------

typedef __attribute__((ext_vector_type(16))) _Float16 v16h;
typedef __attribute__((ext_vector_type(8)))  _Float16 v8h;
typedef __attribute__((ext_vector_type(8)))  float    v8f;
typedef __attribute__((ext_vector_type(4)))  float    v4f;
typedef __attribute__((ext_vector_type(4)))  unsigned int v4u;
typedef __attribute__((ext_vector_type(8)))  int      v8i;
typedef __attribute__((ext_vector_type(4)))  int      v4i;

#define NB   2
#define NT   4096
#define NC   768
#define NH   12
#define ND   64
#define N3C  2304

#define WMMA_F16(a, b, c) \
  __builtin_amdgcn_wmma_f32_16x16x32_f16(false, (a), false, (b), (short)0, (c), false, false)

// Build v16h A/B fragment from two contiguous 8-half chunks.
__device__ __forceinline__ v16h make_frag(v8h lo, v8h hi) {
  v16h r;
#pragma unroll
  for (int i = 0; i < 8; ++i) { r[i] = lo[i]; r[8 + i] = hi[i]; }
  return r;
}

__device__ __forceinline__ v16h frag_from_f32(const float* p) {
  // elements 0..7 at p[0..7] (caller pre-offsets by 8*half), 8..15 at p[16..23]
  v4f a0 = *(const v4f*)(p + 0);
  v4f a1 = *(const v4f*)(p + 4);
  v4f a2 = *(const v4f*)(p + 16);
  v4f a3 = *(const v4f*)(p + 20);
  v16h r;
#pragma unroll
  for (int i = 0; i < 4; ++i) {
    r[i]      = (_Float16)a0[i];
    r[4 + i]  = (_Float16)a1[i];
    r[8 + i]  = (_Float16)a2[i];
    r[12 + i] = (_Float16)a3[i];
  }
  return r;
}

__device__ __forceinline__ v16h frag_from_f16(const _Float16* p) {
  v8h lo = *(const v8h*)(p + 0);
  v8h hi = *(const v8h*)(p + 16);
  return make_frag(lo, hi);
}

// ---------------------------------------------------------------------------
// Tensor Data Mover: 2D f16 tile load Global -> LDS.
// D# packing per CDNA5 ISA ch.8 (group0 128b, group1 256b; groups 2/3 unused
// for 2D tensors -> zero).  This toolchain's builtin takes 6 args:
//   (uint32x4 g0, int32x8 g1, int32x4 g2, int32x4 g3, int32x8 g4, i32 cpol)
// ---------------------------------------------------------------------------
__device__ __forceinline__ void tdm_load_2d_f16(
    unsigned lds_off, unsigned long long gaddr,
    unsigned tensor_d0, unsigned tensor_d1, unsigned stride0,
    unsigned tile_d0, unsigned tile_d1)
{
  v4u g0;
  g0[0] = 1u;                                        // count=1, user, valid
  g0[1] = lds_off;                                   // lds_addr (bytes)
  g0[2] = (unsigned)(gaddr & 0xffffffffu);           // global_addr[31:0]
  g0[3] = (unsigned)((gaddr >> 32) & 0x01ffffffu)    // global_addr[56:32]
        | (2u << 30);                                // type=2 (image)
  v8i g1;
  g1[0] = (int)(1u << 16);                                   // data_size=2B
  g1[1] = (int)((tensor_d0 & 0xffffu) << 16);                // dim0[15:0]
  g1[2] = (int)(((tensor_d0 >> 16) & 0xffffu)                // dim0[31:16]
        |        ((tensor_d1 & 0xffffu) << 16));             // dim1[15:0]
  g1[3] = (int)(((tensor_d1 >> 16) & 0xffffu)                // dim1[31:16]
        |        ((tile_d0 & 0xffffu) << 16));               // tile_dim0
  g1[4] = (int)(tile_d1 & 0xffffu);                          // tile_dim1 (dim2=0)
  g1[5] = (int)stride0;                                      // dim0_stride lo
  g1[6] = 0;                                                 // stride hi / dim1_stride lo
  g1[7] = 0;
  v4i g2 = {0, 0, 0, 0};
  v4i g3 = {0, 0, 0, 0};
  v8i g4 = {0, 0, 0, 0, 0, 0, 0, 0};
  __builtin_amdgcn_tensor_load_to_lds(g0, g1, g2, g3, g4, 0);
}

// ---------------------------------------------------------------------------
// Kernel 1: QKV projection.  One wave per 16x16 tile of [8192 x 2304].
// ---------------------------------------------------------------------------
__global__ __launch_bounds__(256) void qkv_kernel(
    const float* __restrict__ x, const float* __restrict__ Wa,
    const float* __restrict__ ba,
    _Float16* __restrict__ Qh, _Float16* __restrict__ Kh,
    _Float16* __restrict__ Vt)
{
  const int lane = threadIdx.x & 31;
  const int wave = threadIdx.x >> 5;
  const int w    = blockIdx.x * 8 + wave;
  const int tileM = w / (N3C / 16);      // 0..511
  const int tileN = w % (N3C / 16);      // 0..143
  const int lo   = lane & 15;
  const int half = lane >> 4;

  const int m = tileM * 16 + lo;         // A row handled by this lane
  const int n = tileN * 16 + lo;         // B column handled by this lane
  const float* xrow = x  + (size_t)m * NC + 8 * half;
  const float* wrow = Wa + (size_t)n * NC + 8 * half;

  const float bias = ba[n];
  v8f acc;
#pragma unroll
  for (int i = 0; i < 8; ++i) acc[i] = bias;

  for (int k0 = 0; k0 < NC; k0 += 32) {
    __builtin_prefetch(xrow + k0 + 128, 0, 1);   // global_prefetch_b8
    v16h a = frag_from_f32(xrow + k0);
    v16h b = frag_from_f32(wrow + k0);
    acc = WMMA_F16(a, b, acc);
  }

  // Scatter into Q (scaled), K, V^T.  C-fragment: col = lo, row = r + 8*half.
  const int sect = n / NC;               // 0=Q 1=K 2=V
  const int cc   = n % NC;
  const int h    = cc >> 6;
  const int d    = cc & 63;
#pragma unroll
  for (int r = 0; r < 8; ++r) {
    const int row = tileM * 16 + r + 8 * half;
    const int bb  = row >> 12;           // /4096
    const int t   = row & (NT - 1);
    const float v = acc[r];
    if (sect == 0) {
      Qh[((size_t)(bb * NH + h) * NT + t) * ND + d] = (_Float16)(v * 0.125f);
    } else if (sect == 1) {
      Kh[((size_t)(bb * NH + h) * NT + t) * ND + d] = (_Float16)v;
    } else {
      Vt[((size_t)(bb * NH + h) * ND + d) * NT + t] = (_Float16)v;
    }
  }
}

// ---------------------------------------------------------------------------
// Kernel 2: flash attention.  One block = 8 query tiles of one (b,h).
// K/V staged through LDS by the TDM, consumed by all 8 waves.
// ---------------------------------------------------------------------------
#define P_STRIDE 40   // halves; 80B row stride keeps 16B alignment, pads banks

__global__ __launch_bounds__(256) void attn_kernel(
    const _Float16* __restrict__ Qh, const _Float16* __restrict__ Kh,
    const _Float16* __restrict__ Vt, _Float16* __restrict__ Y)
{
  __shared__ _Float16 ldsK[32 * ND];          // [key 0..31][d 0..63]  4KB
  __shared__ _Float16 ldsV[ND * 32];          // [d 0..63][key 0..31]  4KB
  __shared__ _Float16 ldsP[8][16 * P_STRIDE]; // per-wave P transpose  10KB

  const int lane = threadIdx.x & 31;
  const int wave = threadIdx.x >> 5;
  const int bh   = blockIdx.x >> 5;           // 0..23  (b*NH + h)
  const int qt   = (blockIdx.x & 31) * 8 + wave;  // query tile 0..255
  const int lo   = lane & 15;
  const int half = lane >> 4;

  // Q fragments: 16x64 = two 16x32 A-fragments (k = d dimension).
  const _Float16* qrow = Qh + ((size_t)bh * NT + qt * 16 + lo) * ND;
  v16h qf0 = frag_from_f16(qrow + 0  + 8 * half);
  v16h qf1 = frag_from_f16(qrow + 32 + 8 * half);

  float mrow[8], lrow[8];
  v8f o[4];
#pragma unroll
  for (int r = 0; r < 8; ++r) { mrow[r] = -INFINITY; lrow[r] = 0.0f; }
#pragma unroll
  for (int t = 0; t < 4; ++t)
#pragma unroll
    for (int r = 0; r < 8; ++r) o[t][r] = 0.0f;

  _Float16* myP = &ldsP[wave][0];
  const _Float16* kbase = Kh + (size_t)bh * NT * ND;
  const _Float16* vbase = Vt + (size_t)bh * ND * NT;
  const unsigned ldsK_off = (unsigned)(unsigned long long)(uintptr_t)&ldsK[0];
  const unsigned ldsV_off = (unsigned)(unsigned long long)(uintptr_t)&ldsV[0];

  for (int kb = 0; kb < NT / 32; ++kb) {
    const int key0 = kb * 32;

    // ---- TDM: stage K (32x64) and V^T (64x32) tiles into LDS ------------
    if (wave == 0) {
      tdm_load_2d_f16(ldsK_off,
                      (unsigned long long)(uintptr_t)(kbase + (size_t)key0 * ND),
                      /*tensor_d0=*/ND, /*tensor_d1=*/NT, /*stride0=*/ND,
                      /*tile_d0=*/ND, /*tile_d1=*/32);
      tdm_load_2d_f16(ldsV_off,
                      (unsigned long long)(uintptr_t)(vbase + key0),
                      /*tensor_d0=*/NT, /*tensor_d1=*/ND, /*stride0=*/NT,
                      /*tile_d0=*/32, /*tile_d1=*/ND);
      __builtin_amdgcn_s_wait_tensorcnt(0);
    }
    __syncthreads();

    // ---- S = (Q/sqrt(D)) K^T for 32 keys: two 16x16 C tiles -------------
    v8f s[2];
#pragma unroll
    for (int j = 0; j < 2; ++j) {
      const _Float16* krow = &ldsK[(j * 16 + lo) * ND];
      v16h kf0 = frag_from_f16(krow + 0  + 8 * half);
      v16h kf1 = frag_from_f16(krow + 32 + 8 * half);
      v8f c;
#pragma unroll
      for (int r = 0; r < 8; ++r) c[r] = 0.0f;
      c = WMMA_F16(qf0, kf0, c);
      c = WMMA_F16(qf1, kf1, c);
      s[j] = c;
    }

    // ---- online softmax (row = slot r + 8*half, col = lo within half) ---
    float mnew[8], alpha[8], p0[8], p1[8];
#pragma unroll
    for (int r = 0; r < 8; ++r) {
      float mx = fmaxf(s[0][r], s[1][r]);
      mx = fmaxf(mx, __shfl_xor(mx, 8, 32));
      mx = fmaxf(mx, __shfl_xor(mx, 4, 32));
      mx = fmaxf(mx, __shfl_xor(mx, 2, 32));
      mx = fmaxf(mx, __shfl_xor(mx, 1, 32));
      mnew[r]  = fmaxf(mrow[r], mx);
      alpha[r] = __expf(mrow[r] - mnew[r]);
      p0[r]    = __expf(s[0][r] - mnew[r]);
      p1[r]    = __expf(s[1][r] - mnew[r]);
      mrow[r]  = mnew[r];
    }
#pragma unroll
    for (int r = 0; r < 8; ++r) {
      float sum = p0[r] + p1[r];
      sum += __shfl_xor(sum, 8, 32);
      sum += __shfl_xor(sum, 4, 32);
      sum += __shfl_xor(sum, 2, 32);
      sum += __shfl_xor(sum, 1, 32);
      lrow[r] = lrow[r] * alpha[r] + sum;
    }
#pragma unroll
    for (int t = 0; t < 4; ++t)
#pragma unroll
      for (int r = 0; r < 8; ++r) o[t][r] *= alpha[r];

    // ---- C-layout P -> LDS -> A-fragment layout -------------------------
#pragma unroll
    for (int r = 0; r < 8; ++r) {
      const int row = r + 8 * half;
      myP[row * P_STRIDE + lo]      = (_Float16)p0[r];
      myP[row * P_STRIDE + 16 + lo] = (_Float16)p1[r];
    }
    asm volatile("s_wait_dscnt 0" ::: "memory");
    const _Float16* prow = myP + lo * P_STRIDE;
    v16h pf = frag_from_f16(prow + 8 * half);
    asm volatile("" ::: "memory");

    // ---- O += P V  (V^T tile rows are contiguous over keys) -------------
#pragma unroll
    for (int t = 0; t < 4; ++t) {
      const _Float16* vrow = &ldsV[(t * 16 + lo) * 32];
      v16h vf = frag_from_f16(vrow + 8 * half);
      o[t] = WMMA_F16(pf, vf, o[t]);
    }
    __syncthreads();   // protect LDS tiles before next TDM overwrite
  }

  // ---- epilogue: Y[b, t, h*64 + d] = O / l ------------------------------
  const int b = bh / NH;
  const int h = bh % NH;
#pragma unroll
  for (int t = 0; t < 4; ++t) {
#pragma unroll
    for (int r = 0; r < 8; ++r) {
      const int row = qt * 16 + r + 8 * half;
      const float val = o[t][r] / lrow[r];
      Y[((size_t)b * NT + row) * NC + h * 64 + t * 16 + lo] = (_Float16)val;
    }
  }
}

// ---------------------------------------------------------------------------
// Kernel 3: output projection.  One wave per 16x16 tile of [8192 x 768].
// ---------------------------------------------------------------------------
__global__ __launch_bounds__(256) void proj_kernel(
    const _Float16* __restrict__ Yin, const float* __restrict__ Wp,
    const float* __restrict__ bp, float* __restrict__ out)
{
  const int lane = threadIdx.x & 31;
  const int wave = threadIdx.x >> 5;
  const int w    = blockIdx.x * 8 + wave;
  const int tileM = w / (NC / 16);       // 0..511
  const int tileN = w % (NC / 16);       // 0..47
  const int lo   = lane & 15;
  const int half = lane >> 4;

  const int m = tileM * 16 + lo;
  const int n = tileN * 16 + lo;
  const _Float16* yrow = Yin + (size_t)m * NC + 8 * half;
  const float*    wrow = Wp  + (size_t)n * NC + 8 * half;

  const float bias = bp[n];
  v8f acc;
#pragma unroll
  for (int i = 0; i < 8; ++i) acc[i] = bias;

  for (int k0 = 0; k0 < NC; k0 += 32) {
    __builtin_prefetch(yrow + k0 + 128, 0, 1);
    v16h a = frag_from_f16(yrow + k0);
    v16h b = frag_from_f32(wrow + k0);
    acc = WMMA_F16(a, b, acc);
  }

#pragma unroll
  for (int r = 0; r < 8; ++r) {
    const int row = tileM * 16 + r + 8 * half;
    out[(size_t)row * NC + n] = acc[r];
  }
}

// ---------------------------------------------------------------------------
extern "C" void kernel_launch(void* const* d_in, const int* in_sizes, int n_in,
                              void* d_out, int out_size, void* d_ws, size_t ws_size,
                              hipStream_t stream)
{
  (void)in_sizes; (void)n_in; (void)out_size; (void)ws_size;
  const float* x  = (const float*)d_in[0];
  const float* Wa = (const float*)d_in[1];
  const float* ba = (const float*)d_in[2];
  const float* Wp = (const float*)d_in[3];
  const float* bp = (const float*)d_in[4];
  float* out = (float*)d_out;

  const size_t qk_elems = (size_t)NB * NH * NT * ND;   // 6,291,456 f16 each
  char* ws = (char*)d_ws;
  _Float16* Qh = (_Float16*)(ws);
  _Float16* Kh = (_Float16*)(ws + 2 * qk_elems);
  _Float16* Vt = (_Float16*)(ws + 4 * qk_elems);
  _Float16* Y  = (_Float16*)(ws + 6 * qk_elems);

  // 512*144 tiles / 8 waves per block
  qkv_kernel<<<9216, 256, 0, stream>>>(x, Wa, ba, Qh, Kh, Vt);
  // 24 (b,h) * 32 blocks, each block = 8 query tiles
  attn_kernel<<<768, 256, 0, stream>>>(Qh, Kh, Vt, Y);
  // 512*48 tiles / 8 waves per block
  proj_kernel<<<3072, 256, 0, stream>>>(Y, Wp, bp, out);
}
